// TrustGCN_18330920419681
// MI455X (gfx1250) — matrix operand order repaired and checked by reading, not compiled, not run on hardware
//
#include <hip/hip_runtime.h>
#include <hip/hip_bf16.h>
#include <math.h>

// ---------------------------------------------------------------------------
// GCN on MI455X (gfx1250). Memory-bound scatter/gather stays f32 (L2-resident:
// 192MB L2 >> 25.6MB per-node arrays). The six per-node GEMMs use
// v_wmma_f32_16x16x32_f16 with ZERO-PADDED operands so fragment loads are
// unconditional b128 vector loads (no per-element guards / exec masking).
// ---------------------------------------------------------------------------

typedef __attribute__((ext_vector_type(16))) _Float16 v16h;
typedef __attribute__((ext_vector_type(8)))  _Float16 v8h;
typedef __attribute__((ext_vector_type(8)))  float    v8f;

// Build a v16h fragment from two contiguous 16-byte chunks (chunk1 at +8 halves
// for both A's contiguous-k row slice and B's packed layout).
__device__ __forceinline__ v16h frag2(const _Float16* p0, const _Float16* p1) {
  v8h lo = *reinterpret_cast<const v8h*>(p0);
  v8h hi = *reinterpret_cast<const v8h*>(p1);
  v16h r;
#pragma unroll
  for (int e = 0; e < 8; ++e) { r[e] = lo[e]; r[e + 8] = hi[e]; }
  return r;
}

// ---------------- degree / normalization ----------------
__global__ void k_count_deg(const int* __restrict__ dst, unsigned* __restrict__ cnt, int E) {
  int stride = gridDim.x * blockDim.x;
  for (int i = blockIdx.x * blockDim.x + threadIdx.x; i < E; i += stride)
    atomicAdd(&cnt[dst[i]], 1u);
}

__global__ void k_dinv(const unsigned* __restrict__ cnt, float* __restrict__ dinv, int N) {
  int i = blockIdx.x * blockDim.x + threadIdx.x;
  if (i < N) dinv[i] = rsqrtf((float)(cnt[i] + 1u));   // +1 self-loop => deg >= 1
}

// x [N,4] f32 -> ACT [N,32] f16 zero-padded
__global__ void k_cvt_x(const float* __restrict__ x, _Float16* __restrict__ ACT, int N) {
  long long total  = (long long)N << 5;
  long long stride = (long long)gridDim.x * blockDim.x;
  for (long long i = (long long)blockIdx.x * blockDim.x + threadIdx.x; i < total; i += stride) {
    int n = (int)(i >> 5);
    int f = (int)(i & 31);
    ACT[i] = (f < 4) ? (_Float16)x[(size_t)n * 4 + f] : (_Float16)0.0f;
  }
}

// ---------------- pack W [K,F] f32 -> B fragments [ftiles][ktiles][32][16] f16
// B layout (this code's contiguous k order, same as A): lane L holds col N=L%16; element e holds K=16*(L/16)+e.
__global__ void k_pack_w(const float* __restrict__ W, _Float16* __restrict__ Bp,
                         int K, int F, int ktiles, int ftiles) {
  int total  = (ftiles * ktiles) << 9;   // *512
  int stride = gridDim.x * blockDim.x;
  for (int i = blockIdx.x * blockDim.x + threadIdx.x; i < total; i += stride) {
    int e    = i & 15;
    int lane = (i >> 4) & 31;
    int tile = i >> 9;
    int kt   = tile % ktiles;
    int nt   = tile / ktiles;
    int kb   = (kt << 5) + ((lane >> 4) << 4) + e;
    int col  = (nt << 4) + (lane & 15);
    Bp[i] = (kb < K && col < F) ? (_Float16)W[(size_t)kb * F + col] : (_Float16)0.0f;
  }
}

// ---------------- WMMA GEMM: H[N,Fpad] = A[N,Kpad] @ Bp --------------------
// A fragment: lane L row M=rbase+L%16; elems 0..7 = K kk+16*(L/16)..+7,
//             elems 8..15 same +8  -> two b128 loads.
// D: lane L elem v -> row rbase + v + 8*(L/16), col nt*16 + L%16.
// N%16==0, Kpad%32==0, Fpad%16==0: no guards anywhere, EXEC stays full.
__global__ void k_gemm_wmma(const _Float16* __restrict__ A, const _Float16* __restrict__ Bp,
                            float* __restrict__ H, int N, int Kpad, int Fpad, int ftiles) {
  const int lane   = threadIdx.x & 31;
  const int half   = lane >> 4;
  const int rcol   = lane & 15;
  const int wave   = blockIdx.x * (blockDim.x >> 5) + (threadIdx.x >> 5);
  const int nwaves = gridDim.x * (blockDim.x >> 5);
  const int ntiles = N >> 4;
  const int ktiles = Kpad >> 5;

  for (int t = wave; t < ntiles; t += nwaves) {
    const int rbase = t << 4;
    const _Float16* arow = A + (size_t)(rbase + rcol) * Kpad + (half << 4);
    for (int nt = 0; nt < ftiles; ++nt) {
      const _Float16* bptr = Bp + (((size_t)nt * ktiles) << 9) + (lane << 4);
      v8f acc = {};
      for (int kt = 0; kt < ktiles; ++kt) {
        v16h a = frag2(arow + (kt << 5), arow + (kt << 5) + 8);
        v16h b = frag2(bptr + (kt << 9), bptr + (kt << 9) + 8);
        acc = __builtin_amdgcn_wmma_f32_16x16x32_f16(
            false, a, false, b, (short)0, acc, false, false);
      }
      float* out = H + ((size_t)rbase + (half << 3)) * Fpad + (nt << 4) + rcol;
#pragma unroll
      for (int v = 0; v < 8; ++v) out[(size_t)v * Fpad] = acc[v];
    }
  }
}

// ---------------- edge scatter: AGG[dst] += H[src] * dinv[src]*dinv[dst] -----
__global__ void k_scatter(const int* __restrict__ src, const int* __restrict__ dst,
                          const float* __restrict__ dinv, const float* __restrict__ H,
                          float* __restrict__ AGG, int E, int F, int gshift) {
  long long total  = (long long)E << gshift;
  long long stride = (long long)gridDim.x * blockDim.x;
  int gmask = (1 << gshift) - 1;
  for (long long i = (long long)blockIdx.x * blockDim.x + threadIdx.x; i < total; i += stride) {
    long long ip = i + (stride << 2);              // stream-ahead prefetch of edge list
    if (ip < total) {
      int ep = (int)(ip >> gshift);
      __builtin_prefetch(&src[ep], 0, 1);
      __builtin_prefetch(&dst[ep], 0, 1);
    }
    int e = (int)(i >> gshift);
    int g = (int)(i & gmask);
    int s = src[e], d = dst[e];
    float nrm = dinv[s] * dinv[d];
    const float4 hv = *reinterpret_cast<const float4*>(H + (size_t)s * F + (g << 2));
    float* o = AGG + (size_t)d * F + (g << 2);
    atomicAdd(o + 0, hv.x * nrm);
    atomicAdd(o + 1, hv.y * nrm);
    atomicAdd(o + 2, hv.z * nrm);
    atomicAdd(o + 3, hv.w * nrm);
  }
}

// ---- fused self-loop + bias + ELU -> f16 activations, zero-padded to Kpad ---
__global__ void k_finalize_gcn(const float* __restrict__ AGG, const float* __restrict__ H,
                               const float* __restrict__ dinv, const float* __restrict__ b,
                               _Float16* __restrict__ ACT, int N, int F, int kpshift) {
  int Kpad = 1 << kpshift;
  long long total  = (long long)N << kpshift;
  long long stride = (long long)gridDim.x * blockDim.x;
  for (long long i = (long long)blockIdx.x * blockDim.x + threadIdx.x; i < total; i += stride) {
    int n = (int)(i >> kpshift);
    int f = (int)(i & (Kpad - 1));
    _Float16 o = (_Float16)0.0f;
    if (f < F) {
      float di = dinv[n];
      size_t j = (size_t)n * F + f;
      float v  = AGG[j] + H[j] * di * di + b[f];     // self-loop norm = dinv^2
      o = (_Float16)(v > 0.0f ? v : expm1f(v));      // ELU(alpha=1)
    }
    ACT[i] = o;
  }
}

__global__ void k_bias_elu(const float* __restrict__ Z, const float* __restrict__ b,
                           _Float16* __restrict__ ACT, int N, int F, int kpshift) {
  int Kpad = 1 << kpshift;
  long long total  = (long long)N << kpshift;
  long long stride = (long long)gridDim.x * blockDim.x;
  for (long long i = (long long)blockIdx.x * blockDim.x + threadIdx.x; i < total; i += stride) {
    int n = (int)(i >> kpshift);
    int f = (int)(i & (Kpad - 1));
    _Float16 o = (_Float16)0.0f;
    if (f < F) {
      float v = Z[(size_t)n * F + f] + b[f];
      o = (_Float16)(v > 0.0f ? v : expm1f(v));
    }
    ACT[i] = o;
  }
}

// ---------------- 2-class log-softmax head (Z has row stride 16) -------------
__global__ void k_head(const float* __restrict__ Z, const float* __restrict__ b,
                       float* __restrict__ OUT, int N) {
  int n = blockIdx.x * blockDim.x + threadIdx.x;
  if (n < N) {
    float a0 = Z[(size_t)n * 16 + 0] + b[0];
    float a1 = Z[(size_t)n * 16 + 1] + b[1];
    float m   = fmaxf(a0, a1);
    float lse = m + logf(expf(a0 - m) + expf(a1 - m));
    OUT[2 * n + 0] = a0 - lse;
    OUT[2 * n + 1] = a1 - lse;
  }
}

// ---------------------------------------------------------------------------
extern "C" void kernel_launch(void* const* d_in, const int* in_sizes, int n_in,
                              void* d_out, int out_size, void* d_ws, size_t ws_size,
                              hipStream_t stream) {
  const float* x   = (const float*)d_in[0];
  const int*   ei  = (const int*)d_in[1];        // [2, E] int32
  const float* W1  = (const float*)d_in[2];  const float* b1  = (const float*)d_in[3];
  const float* W2  = (const float*)d_in[4];  const float* b2  = (const float*)d_in[5];
  const float* W3  = (const float*)d_in[6];  const float* b3  = (const float*)d_in[7];
  const float* Wf1 = (const float*)d_in[8];  const float* bf1 = (const float*)d_in[9];
  const float* Wf2 = (const float*)d_in[10]; const float* bf2 = (const float*)d_in[11];
  const float* Wf3 = (const float*)d_in[12]; const float* bf3 = (const float*)d_in[13];

  const int N = in_sizes[0] / 4;     // 100000 (multiple of 16)
  const int E = in_sizes[1] / 2;     // 3200000
  const int* src = ei;
  const int* dst = ei + E;

  // workspace carve-out (every region fully written before read)
  auto al = [](size_t v) { return (v + 255) & ~(size_t)255; };
  char* p = (char*)d_ws;
  unsigned* cnt  = (unsigned*)p;  p += al((size_t)N * 4);
  float*    dinv = (float*)p;     p += al((size_t)N * 4);
  _Float16* act  = (_Float16*)p;  p += al((size_t)N * 64 * 2);   // [N,Kpad<=64]
  float*    hbuf = (float*)p;     p += al((size_t)N * 64 * 4);   // [N,Fpad<=64]
  float*    agg  = (float*)p;     p += al((size_t)N * 64 * 4);
  _Float16* bp   = (_Float16*)p;  p += al((size_t)8 * 512 * 2);  // packed B frags

  const int T = 256;
  const int gblocks = ((N >> 4) + 7) / 8;   // 8 waves/block, one 16-row tile/wave

  // deg = in-degree + self-loop; dinv = rsqrt(deg)
  hipMemsetAsync(cnt, 0, (size_t)N * 4, stream);
  k_count_deg<<<2048, T, 0, stream>>>(dst, cnt, E);
  k_dinv<<<(N + T - 1) / T, T, 0, stream>>>(cnt, dinv, N);

  // x -> padded f16 activations [N,32]
  k_cvt_x<<<2048, T, 0, stream>>>(x, act, N);

  auto gemm = [&](const float* W, int K, int F, int Kpad, int Fpad) {
    int ktiles = Kpad >> 5, ftiles = Fpad >> 4;
    int ptotal = (ftiles * ktiles) << 9;
    k_pack_w<<<(ptotal + T - 1) / T, T, 0, stream>>>(W, bp, K, F, ktiles, ftiles);
    k_gemm_wmma<<<gblocks, T, 0, stream>>>(act, bp, hbuf, N, Kpad, Fpad, ftiles);
  };

  auto gcn = [&](const float* W, const float* b, int K, int F, int Kpad,
                 int fshift, int kpshift_next) {
    gemm(W, K, F, Kpad, F);                       // Fpad == F (16/32/64)
    hipMemsetAsync(agg, 0, (size_t)N * F * 4, stream);
    k_scatter<<<4096, T, 0, stream>>>(src, dst, dinv, hbuf, agg, E, F, fshift - 2);
    k_finalize_gcn<<<4096, T, 0, stream>>>(agg, hbuf, dinv, b, act, N, F, kpshift_next);
  };
  gcn(W1, b1,  4, 16, 32, 4, 5);   // ACT[N,32] (16 real + pad)
  gcn(W2, b2, 16, 32, 32, 5, 5);   // ACT[N,32]
  gcn(W3, b3, 32, 64, 32, 6, 6);   // ACT[N,64]

  auto mlp = [&](const float* W, const float* b, int K, int F, int Kpad, int kpshift_next) {
    gemm(W, K, F, Kpad, F);
    k_bias_elu<<<4096, T, 0, stream>>>(hbuf, b, act, N, F, kpshift_next);
  };
  mlp(Wf1, bf1, 64, 32, 64, 5);    // ACT[N,32]
  mlp(Wf2, bf2, 32, 16, 32, 5);    // ACT[N,32] (16 real + pad)

  gemm(Wf3, 16, 2, 32, 16);        // Fpad=16, cols 2..15 compute to zero
  k_head<<<(N + T - 1) / T, T, 0, stream>>>(hbuf, bf3, (float*)d_out, N);
}
